// FastRCNN_17128329576558
// MI455X (gfx1250) — matrix operand order repaired and checked by reading, not compiled, not run on hardware
//
#include <hip/hip_runtime.h>
#include <hip/hip_bf16.h>

// ---------------------------------------------------------------------------
// Fast R-CNN (VGG16-BN backbone, stride 16) for MI455X / gfx1250, wave32.
// GEMM-like work (convs via implicit im2col, FC layers) runs on
// v_wmma_f32_16x16x32_bf16 with f32 accumulation. bf16 keeps the 103M-param
// fc1 weight matrix (205 MB) ~L2-resident (192 MB L2) across M-tile passes.
//
// LDS tiles are FRAGMENT-MAJOR (each lane's 16-half fragment is 32 contiguous
// bytes -> two ds_load_b128). Conv uses TAP-MAJOR K ordering (k = tap*Cin+ci)
// so the 3x3 tap, halo test and base address are K-tile-invariant.
//
// CDNA5 data movement:
//  - GEMM A tiles: GLOBAL_LOAD_ASYNC_TO_LDS_B32 per-lane scatter (ASYNCcnt).
//  - GEMM B panels: TENSOR_LOAD_TO_LDS (TDM) with an SGPR descriptor,
//    double-buffered linear landing zone (s_wait_tensorcnt 1 overlap),
//    then an LDS->LDS repack into the fragment-major swizzle.
// ---------------------------------------------------------------------------

#define USE_ASYNC_LDS 1
#define USE_TDM 1

typedef __bf16 bf16;
typedef __attribute__((ext_vector_type(16))) __bf16 v16bf;
typedef __attribute__((ext_vector_type(8)))  float  v8f;
typedef __attribute__((ext_vector_type(4)))  unsigned v4u;
typedef __attribute__((ext_vector_type(8)))  unsigned v8u;

union H2 { bf16 h[2]; unsigned u; };        // packed bf16 pair
struct Q2 { uint4 a, b; };                  // 32B: one v16bf fragment

static __device__ __forceinline__ v16bf load_frag(const unsigned* p) {
  Q2 q = *reinterpret_cast<const Q2*>(p);   // 2x ds_load_b128
  return __builtin_bit_cast(v16bf, q);
}
static __device__ __forceinline__ int imax(int a, int b) { return a > b ? a : b; }
static __device__ __forceinline__ int imin(int a, int b) { return a < b ? a : b; }

#if USE_ASYNC_LDS
// 4B global -> LDS async copy, per-lane scatter (GLOBAL_LOAD_ASYNC_TO_LDS_B32)
static __device__ __forceinline__ void async_g2l_b32(unsigned lds_byte_addr,
                                                     const void* gaddr) {
  asm volatile("global_load_async_to_lds_b32 %0, %1, off"
               :: "v"(lds_byte_addr), "v"(gaddr) : "memory");
}
static __device__ __forceinline__ void async_wait0() {
  asm volatile("s_wait_asynccnt 0x0" ::: "memory");
}
#endif

#if USE_TDM
// TDM: DMA a [rows x 128] bf16 2D tile (row stride = N elements) into LDS.
// D# per CDNA5 ISA 8.3/8.4; groups in SGPRs, VADDR2/3 omitted (2D tensor).
static __device__ __forceinline__ void tdm_load_2d(unsigned lds_byte_addr,
                                                   const void* gaddr,
                                                   unsigned N, unsigned K) {
  unsigned long long ga = (unsigned long long)gaddr;
  v4u g0;
  g0[0] = 1u;                                    // count=1, user descriptor
  g0[1] = lds_byte_addr;                         // lds_addr [63:32]
  g0[2] = (unsigned)ga;                          // global_addr lo
  g0[3] = (unsigned)((ga >> 32) & 0x01FFFFFFu)   // global_addr [56:32]
        | (2u << 30);                            // type=2 ("image")
  v8u g1;
  g1[0] = 1u << 16;                              // wg_mask=0, data_size=2B
  g1[1] = (N & 0xFFFFu) << 16;                   // tensor_dim0[15:0] @bit48
  g1[2] = ((N >> 16) & 0xFFFFu)                  // tensor_dim0[31:16]
        | ((K & 0xFFFFu) << 16);                 // tensor_dim1[15:0]
  g1[3] = ((K >> 16) & 0xFFFFu)                  // tensor_dim1[31:16]
        | (128u << 16);                          // tile_dim0 = 128
  g1[4] = 32u;                                   // tile_dim1 = 32, tile_dim2=0
  g1[5] = N;                                     // tensor_dim0_stride lo32
  g1[6] = 0u;                                    // stride hi, dim1_stride lo
  g1[7] = 0u;
  asm volatile("tensor_load_to_lds %0, %1" :: "s"(g0), "s"(g1) : "memory");
}
#endif

// ---------------------------------------------------------------------------
// fp32 -> bf16 conversion (input image + FC weights)
// ---------------------------------------------------------------------------
__global__ void k_f32_to_bf16(const float* __restrict__ in,
                              bf16* __restrict__ out, int n) {
  int i = blockIdx.x * blockDim.x + threadIdx.x;
  if (i < n) out[i] = (bf16)in[i];
}

// ---------------------------------------------------------------------------
// 3x3 SAME conv + bias + BN + ReLU as implicit GEMM on WMMA.
//   M = Cout (16/block), K = Cin*9 (step 32), N = B*H*W (128/block, 16/wave)
// FIRST=true : layer 0 (Cin=3, K=27): ci-major K with clamped+selected loads.
// FIRST=false: Cin = 1<<cs (64..512): TAP-MAJOR K, branchless staging.
// ---------------------------------------------------------------------------
template <bool FIRST>
__global__ __launch_bounds__(256) void k_conv3x3_bn_relu(
    const bf16* __restrict__ in,    // [B,Cin,H,W] bf16
    const float* __restrict__ wgt,  // [Cout][Cin][3][3] fp32
    const float* __restrict__ bias,
    const float* __restrict__ gamma,
    const float* __restrict__ beta,
    const float* __restrict__ mean,
    const float* __restrict__ var,
    bf16* __restrict__ out,         // [B,Cout,H,W] bf16
    int NB, int Cin, int cs, int Cout, int H, int W) {
  const int Ktot = Cin * 9;
  const int HW   = H * W;
  const int NPIX = NB * HW;
  const int p0   = blockIdx.x * 128;
  const int c0   = blockIdx.y * 16;
  const int tid  = threadIdx.x;
  const int lane = tid & 31;
  const int wv   = tid >> 5;           // 8 waves, each owns 16 pixels

  __shared__ __align__(16) unsigned sA[16 * 16];      // [32 lanes][8 words]
  __shared__ __align__(16) unsigned sB[8 * 32 * 8];   // [8 grp][32 lanes][8 w]

  // hoisted pixel decomposition for B staging (n fixed per thread)
  const int nB   = tid & 127;
  const int kpB0 = tid >> 7;           // 0..1
  const int p    = p0 + nB;
  const bool pok = p < NPIX;
  int pb = 0, py = 0, px = 0;
  if (pok) { pb = p / HW; int rem = p % HW; py = rem / W; px = rem % W; }
  const unsigned bAddrBase =
      (unsigned)(((nB >> 4) * 32 + (nB & 15)) * 8);

  // A staging coords (one bf16 pair per thread)
  const int mA  = tid >> 4;            // 0..15
  const int kpA = tid & 15;            // k = 2*kpA
  const unsigned aAddr =
      (unsigned)(((mA & 15) + 16 * (kpA >> 3)) * 8 + (kpA & 7));
  const float* wrow = wgt + (long)(c0 + mA) * Ktot;

  v8f acc = {};

  for (int k0 = 0; k0 < Ktot; k0 += 32) {
    // ---------------- stage A (weights, fp32 -> bf16 pair) ----------------
    {
      H2 pk;
      float v0, v1;
      if (FIRST) {
        int gk = k0 + 2 * kpA;
        v0 = wrow[imin(gk, Ktot - 1)];
        v1 = wrow[imin(gk + 1, Ktot - 1)];
        v0 = (gk < Ktot) ? v0 : 0.f;
        v1 = (gk + 1 < Ktot) ? v1 : 0.f;
      } else {
        // tap-major: k = tap*Cin + ci  ->  w[m][ci][tap], stride-9 gather
        int tap = k0 >> cs;
        int ci  = (k0 & (Cin - 1)) + 2 * kpA;
        v0 = wrow[(long)ci * 9 + tap];
        v1 = wrow[(long)(ci + 1) * 9 + tap];
      }
      pk.h[0] = (bf16)v0; pk.h[1] = (bf16)v1;
      sA[aAddr] = pk.u;
    }
    // ---------------- stage B (implicit im2col, branchless) ---------------
    if (FIRST) {
#pragma unroll
      for (int t = 0; t < 8; ++t) {
        int kp = kpB0 + t * 2;
        int gk = k0 + 2 * kp;
        H2 pk;
#pragma unroll
        for (int e = 0; e < 2; ++e) {
          int gke = imin(gk + e, Ktot - 1);
          int ci = gke / 9, off = gke % 9;
          int yy = py + off / 3 - 1, xx = px + off % 3 - 1;
          bool ok = pok && (gk + e < Ktot) &&
                    (unsigned)yy < (unsigned)H && (unsigned)xx < (unsigned)W;
          long a = ok ? (((long)(pb * Cin + ci) * H + yy) * W + xx) : 0;
          bf16 v = in[a];
          pk.h[e] = ok ? v : (bf16)0.f;
        }
        sB[bAddrBase + (unsigned)(16 * (kp >> 3)) * 8 + (kp & 7)] = pk.u;
      }
    } else {
      // tap and halo test are K-tile-invariant; ci is linear
      int tap = k0 >> cs;
      int dy = tap / 3 - 1, dx = tap % 3 - 1;
      int ci0 = k0 & (Cin - 1);
      int yy = py + dy, xx = px + dx;
      bool ok = pok && (unsigned)yy < (unsigned)H && (unsigned)xx < (unsigned)W;
      long base = ok ? (((long)((pb << cs) + ci0) * H + yy) * W + xx) : 0;
#pragma unroll
      for (int t = 0; t < 8; ++t) {
        int kp = kpB0 + t * 2;
        bf16 r0 = in[base + (long)(2 * kp) * HW];
        bf16 r1 = in[base + (long)(2 * kp + 1) * HW];
        H2 pk;
        pk.h[0] = ok ? r0 : (bf16)0.f;
        pk.h[1] = ok ? r1 : (bf16)0.f;
        sB[bAddrBase + (unsigned)(16 * (kp >> 3)) * 8 + (kp & 7)] = pk.u;
      }
    }
    __syncthreads();

    if (!FIRST && k0 + 32 < Ktot)  // prefetch next input-channel slab
      __builtin_prefetch(&in[((long)((pb << cs) + (k0 & (Cin - 1))) + 32) * HW], 0, 1);

    // ---- fragments: 2x ds_load_b128 each ----
    v16bf af  = load_frag(&sA[lane * 8]);
    v16bf bfg = load_frag(&sB[(wv * 32 + lane) * 8]);
    acc = __builtin_amdgcn_wmma_f32_16x16x32_bf16(
        false, af, false, bfg, (short)0, acc, false, false);
    __syncthreads();
  }

  // ---- epilogue: bias + BN + ReLU, bf16 store ----
  const int nloc = (lane & 15) + wv * 16;
  const int pe = p0 + nloc;
  if (pe < NPIX) {
    const int b = pe / HW, rem = pe % HW;
#pragma unroll
    for (int r = 0; r < 8; ++r) {
      int c = c0 + r + 8 * (lane >> 4);
      float s = gamma[c] * rsqrtf(var[c] + 1e-5f);
      float v = (acc[r] + bias[c] - mean[c]) * s + beta[c];
      v = fmaxf(v, 0.f);
      out[((long)b * Cout + c) * HW + rem] = (bf16)v;
    }
  }
}

// ---------------------------------------------------------------------------
// 2x2 stride-2 maxpool (bf16)
// ---------------------------------------------------------------------------
__global__ void k_maxpool2(const bf16* __restrict__ in, bf16* __restrict__ out,
                           int NC, int H, int W) {
  int Ho = H >> 1, Wo = W >> 1;
  int i = blockIdx.x * blockDim.x + threadIdx.x;
  if (i >= NC * Ho * Wo) return;
  int x = i % Wo, t = i / Wo;
  int y = t % Ho, nc = t / Ho;
  const bf16* p = in + ((long)nc * H + 2 * y) * W + 2 * x;
  float a = fmaxf((float)p[0], (float)p[1]);
  float b = fmaxf((float)p[W], (float)p[W + 1]);
  out[i] = (bf16)fmaxf(a, b);
}

// ---------------------------------------------------------------------------
// ROI adaptive max-pool 7x7 over [8,512,14,14] -> pooled [512, 25088] bf16,
// fc-ready layout: row r, column c*49 + i*7 + j.
// ---------------------------------------------------------------------------
__global__ void k_roi_pool(const bf16* __restrict__ feat,
                           const float* __restrict__ rois,
                           const int* __restrict__ roi_idx,
                           bf16* __restrict__ pooled) {
  const int H = 14, W = 14, C = 512, O = 7, R = 512;
  int idx = blockIdx.x * blockDim.x + threadIdx.x;
  if (idx >= R * C * O * O) return;
  int j = idx % O; int t = idx / O;
  int i = t % O;  t /= O;
  int c = t % C;  int r = t / C;
  int b = roi_idx[r];
  int x1 = (int)floorf(rois[r * 4 + 0] * (float)W);
  int y1 = (int)floorf(rois[r * 4 + 1] * (float)H);
  int x2 = (int)ceilf (rois[r * 4 + 2] * (float)W);
  int y2 = (int)ceilf (rois[r * 4 + 3] * (float)H);
  int Sx = imax(x2 - x1, 1), Sy = imax(y2 - y1, 1);
  int xs = x1 + (j * Sx) / O,  xe = x1 + ((j + 1) * Sx + O - 1) / O;
  int ys = y1 + (i * Sy) / O,  ye = y1 + ((i + 1) * Sy + O - 1) / O;
  xs = imax(xs, 0); ys = imax(ys, 0); xe = imin(xe, W); ye = imin(ye, H);
  float m = -3.402823466e+38f;
  const bf16* f = feat + ((long)b * C + c) * (H * W);
  for (int y = ys; y < ye; ++y)
    for (int x = xs; x < xe; ++x)
      m = fmaxf(m, (float)f[y * W + x]);
  pooled[(long)r * 25088 + c * 49 + i * 7 + j] = (bf16)m;
}

// ---------------------------------------------------------------------------
// bf16 GEMM + bias (+ReLU): C[M,N] = A[M,K] * B[K,N]
// Block tile 32(M) x 128(N), 8 waves as 2x4, each wave two 16x16 WMMAs
// sharing one A fragment. A tile: async global->LDS scatter (ASYNCcnt).
// B panel: TDM tensor_load_to_lds into a double-buffered linear landing
// zone (TENSORcnt, tile i+1 in flight while tile i repacks), then LDS->LDS
// repack into the fragment-major swizzle. TDM zero-fills beyond tensor_dim0,
// covering the N=21/84 heads; NBND adds a belt-and-braces select + store mask.
// ---------------------------------------------------------------------------
template <bool NBND>
__global__ __launch_bounds__(256) void k_gemm_bf16(
    const bf16* __restrict__ A, const bf16* __restrict__ Bw,
    const float* __restrict__ bias, float* __restrict__ outF,
    bf16* __restrict__ outH, int M, int N, int K, int relu) {
  const int nt   = blockIdx.x * 128;
  const int mt   = blockIdx.y * 32;
  const int tid  = threadIdx.x;
  const int lane = tid & 31;
  const int wv   = tid >> 5;
  const int mw   = wv >> 2;            // 0..1
  const int nw   = wv & 3;             // 0..3

  __shared__ __align__(16) unsigned sA[2 * 32 * 8];   // [2 mgrp][32 lanes][8 w]
  __shared__ __align__(16) unsigned sB[8 * 32 * 8];   // [8 grp][32 lanes][8 w]
#if USE_TDM
  __shared__ __align__(16) bf16 sLin[2][32 * 128];    // TDM landing zones
#endif

  // B coords (n fixed per thread)
  const int nB   = tid & 127;
  const int kpB0 = tid >> 7;
  const int gnB  = nt + nB;
  const bool nok = !NBND || gnB < N;
  const int gnBs = NBND ? imin(gnB, N - 1) : gnB;     // clamped, branchless
  const unsigned bAddrBase = (unsigned)(((nB >> 4) * 32 + (nB & 15)) * 8);

  v8f acc0 = {}, acc1 = {};

  const int ntiles = K >> 5;
#if USE_TDM
  if (wv == 0)  // prime the pipeline: tile 0 -> buffer 0
    tdm_load_2d((unsigned)(unsigned long long)&sLin[0][0],
                Bw + (long)0 * N + nt, (unsigned)N, (unsigned)K);
#endif

  for (int it = 0; it < ntiles; ++it) {
    const int k0 = it << 5;
    // ---- stage A: 4B bf16 pair, async global->LDS scatter into swizzle ----
#pragma unroll
    for (int t = 0; t < 2; ++t) {
      int pi = tid + t * 256;          // 0..511
      int m = pi >> 4, kp = pi & 15;
      unsigned aIdx = (unsigned)((m >> 4) * 256 +
          ((m & 15) + 16 * (kp >> 3)) * 8 + (kp & 7));
      const bf16* ga = A + (long)(mt + m) * K + k0 + 2 * kp;
#if USE_ASYNC_LDS
      async_g2l_b32((unsigned)(unsigned long long)&sA[aIdx], ga);
#else
      sA[aIdx] = *(const unsigned*)ga;
#endif
    }

#if USE_TDM
    const int buf = it & 1;
    if (wv == 0) {
      if (it + 1 < ntiles) {           // overlap: DMA tile i+1 while i repacks
        tdm_load_2d((unsigned)(unsigned long long)&sLin[buf ^ 1][0],
                    Bw + (long)(k0 + 32) * N + nt, (unsigned)N, (unsigned)K);
        __builtin_amdgcn_s_wait_tensorcnt(1);   // tile i landed (in-order)
      } else {
        __builtin_amdgcn_s_wait_tensorcnt(0);   // last tile landed
      }
    }
#if USE_ASYNC_LDS
    async_wait0();                     // A-tile DMA complete before barrier
#endif
    __syncthreads();                   // sA + sLin[buf] ready

    // ---- repack linear TDM tile -> fragment-major swizzle (LDS->LDS) ----
#pragma unroll
    for (int t = 0; t < 8; ++t) {
      int kp = kpB0 + t * 2;           // 0..15
      bf16 r0 = sLin[buf][(2 * kp) * 128 + nB];
      bf16 r1 = sLin[buf][(2 * kp + 1) * 128 + nB];
      H2 pk;
      pk.h[0] = nok ? r0 : (bf16)0.f;
      pk.h[1] = nok ? r1 : (bf16)0.f;
      sB[bAddrBase + (unsigned)(16 * (kp >> 3)) * 8 + (kp & 7)] = pk.u;
    }
    __syncthreads();                   // sB ready
#else
    // ---- direct staging fallback: two coalesced row loads per pair ----
#pragma unroll
    for (int t = 0; t < 8; ++t) {
      int kp = kpB0 + t * 2;
      int gk = k0 + 2 * kp;
      bf16 r0 = Bw[(long)gk * N + gnBs];
      bf16 r1 = Bw[(long)(gk + 1) * N + gnBs];
      H2 pk;
      pk.h[0] = nok ? r0 : (bf16)0.f;
      pk.h[1] = nok ? r1 : (bf16)0.f;
      sB[bAddrBase + (unsigned)(16 * (kp >> 3)) * 8 + (kp & 7)] = pk.u;
    }
#if USE_ASYNC_LDS
    async_wait0();
#endif
    __syncthreads();
#endif

    // ---- fragments: 2x ds_load_b128 each ----
    v16bf af = load_frag(&sA[(mw * 32 + lane) * 8]);
    v16bf b0 = load_frag(&sB[((nw * 2 + 0) * 32 + lane) * 8]);
    v16bf b1 = load_frag(&sB[((nw * 2 + 1) * 32 + lane) * 8]);
    acc0 = __builtin_amdgcn_wmma_f32_16x16x32_bf16(
        false, af, false, b0, (short)0, acc0, false, false);
    acc1 = __builtin_amdgcn_wmma_f32_16x16x32_bf16(
        false, af, false, b1, (short)0, acc1, false, false);
    __syncthreads();                   // frag reads done; safe to restage
  }

#pragma unroll
  for (int half = 0; half < 2; ++half) {
    int gn = nt + nw * 32 + (lane & 15) + half * 16;
    if (!NBND || gn < N) {
#pragma unroll
      for (int r = 0; r < 8; ++r) {
        int gm = mt + mw * 16 + r + 8 * (lane >> 4);
        float v = (half ? acc1[r] : acc0[r]) + bias[gn];
        if (relu) v = fmaxf(v, 0.f);
        if (outF) outF[(long)gm * N + gn] = v;
        else      outH[(long)gm * N + gn] = (bf16)v;
      }
    }
  }
}

// ---------------------------------------------------------------------------
// Host orchestration
// ---------------------------------------------------------------------------
extern "C" void kernel_launch(void* const* d_in, const int* in_sizes, int n_in,
                              void* d_out, int out_size, void* d_ws, size_t ws_size,
                              hipStream_t stream) {
  (void)in_sizes; (void)n_in; (void)out_size; (void)ws_size;

  const float* x      = (const float*)d_in[0];
  const float* rois   = (const float*)d_in[1];
  const int*   roiidx = (const int*)d_in[2];

  const int NL = 13;
  static const int  couts[NL]      = {64, 64, 128, 128, 256, 256, 256, 512, 512, 512, 512, 512, 512};
  static const bool pool_after[NL] = {false, true, false, true, false, false, true,
                                      false, false, true, false, false, false};
  const int fcb = 3 + 6 * NL;  // 81

  const int B = 8, R = 512;

  char* ws = (char*)d_ws;
  size_t off = 0;
  auto alloc = [&](size_t bytes) -> void* {
    void* p = ws + off;
    off += (bytes + 255) & ~(size_t)255;
    return p;
  };

  bf16* x_bf   = (bf16*)alloc((size_t)B * 3 * 224 * 224 * 2);
  size_t actsz = (size_t)B * 64 * 224 * 224 * 2;
  bf16* bufA   = (bf16*)alloc(actsz);
  bf16* bufB   = (bf16*)alloc(actsz);
  bf16* pooled = (bf16*)alloc((size_t)R * 25088 * 2);
  bf16* h1     = (bf16*)alloc((size_t)R * 4096 * 2);
  bf16* h2     = (bf16*)alloc((size_t)R * 4096 * 2);
  bf16* w1     = (bf16*)alloc((size_t)25088 * 4096 * 2);  // ~L2-resident bf16
  bf16* w2     = (bf16*)alloc((size_t)4096 * 4096 * 2);
  bf16* wc     = (bf16*)alloc((size_t)4096 * 21 * 2);
  bf16* wb     = (bf16*)alloc((size_t)4096 * 84 * 2);

  auto cvt = [&](const void* src, bf16* dst, long n) {
    k_f32_to_bf16<<<dim3((unsigned)((n + 255) / 256)), 256, 0, stream>>>(
        (const float*)src, dst, (int)n);
  };

  cvt(x, x_bf, (long)B * 3 * 224 * 224);
  cvt(d_in[fcb + 0], w1, (long)25088 * 4096);
  cvt(d_in[fcb + 2], w2, (long)4096 * 4096);
  cvt(d_in[fcb + 4], wc, (long)4096 * 21);
  cvt(d_in[fcb + 6], wb, (long)4096 * 84);

  // ---- VGG16-BN backbone ----
  int C = 3, H = 224, W = 224;
  const bf16* cur = x_bf;
  for (int l = 0; l < NL; ++l) {
    int Cout = couts[l];
    const float* Wc = (const float*)d_in[3 + 6 * l + 0];
    const float* bc = (const float*)d_in[3 + 6 * l + 1];
    const float* gc = (const float*)d_in[3 + 6 * l + 2];
    const float* be = (const float*)d_in[3 + 6 * l + 3];
    const float* mc = (const float*)d_in[3 + 6 * l + 4];
    const float* vc = (const float*)d_in[3 + 6 * l + 5];
    int cshift = 0;
    while ((1 << cshift) < C) ++cshift;  // log2(Cin), exact for l>0
    bf16* dst = (cur == bufA) ? bufB : bufA;
    int npix = B * H * W;
    dim3 grid((npix + 127) / 128, (Cout + 15) / 16);
    if (l == 0)
      k_conv3x3_bn_relu<true><<<grid, 256, 0, stream>>>(
          cur, Wc, bc, gc, be, mc, vc, dst, B, C, cshift, Cout, H, W);
    else
      k_conv3x3_bn_relu<false><<<grid, 256, 0, stream>>>(
          cur, Wc, bc, gc, be, mc, vc, dst, B, C, cshift, Cout, H, W);
    cur = dst;
    C = Cout;
    if (pool_after[l]) {
      bf16* dst2 = (cur == bufA) ? bufB : bufA;
      int n = B * C * (H / 2) * (W / 2);
      k_maxpool2<<<(n + 255) / 256, 256, 0, stream>>>(cur, dst2, B * C, H, W);
      cur = dst2;
      H /= 2; W /= 2;
    }
  }

  // ---- ROI adaptive max-pool ----
  {
    int n = R * 512 * 49;
    k_roi_pool<<<(n + 255) / 256, 256, 0, stream>>>(cur, rois, roiidx, pooled);
  }

  // ---- FC head ----
  {
    dim3 g(4096 / 128, R / 32);
    k_gemm_bf16<false><<<g, 256, 0, stream>>>(pooled, w1, (const float*)d_in[fcb + 1],
                                              nullptr, h1, R, 4096, 25088, 1);
  }
  {
    dim3 g(4096 / 128, R / 32);
    k_gemm_bf16<false><<<g, 256, 0, stream>>>(h1, w2, (const float*)d_in[fcb + 3],
                                              nullptr, h2, R, 4096, 4096, 1);
  }
  float* outF = (float*)d_out;
  {
    dim3 g(1, R / 32);  // cls_score [512,21]
    k_gemm_bf16<true><<<g, 256, 0, stream>>>(h2, wc, (const float*)d_in[fcb + 5],
                                             outF, nullptr, R, 21, 4096, 0);
  }
  {
    dim3 g(1, R / 32);  // bbox [512,84] directly after cls in d_out
    k_gemm_bf16<true><<<g, 256, 0, stream>>>(h2, wb, (const float*)d_in[fcb + 7],
                                             outF + (long)512 * 21, nullptr, R, 84, 4096, 0);
  }
}
